// PEFullyConnected_57801669869758
// MI455X (gfx1250) — matrix operand order repaired and checked by reading, not compiled, not run on hardware
//
#include <hip/hip_runtime.h>

// ---------------- compile-time S4 group tables (match itertools order) -----
struct QTab {
  signed char perms[24][4];   // lexicographic permutations of (0,1,2,3)
  signed char q[6][4][24];    // q[y][s][j]: out += M[b, j, q[y,s,j]]
};

constexpr QTab make_qtab() {
  QTab T{};
  int idx = 0;
  for (int a = 0; a < 4; ++a)
    for (int b = 0; b < 4; ++b) {
      if (b == a) continue;
      for (int c = 0; c < 4; ++c) {
        if (c == a || c == b) continue;
        int d = 6 - a - b - c;
        T.perms[idx][0] = (signed char)a; T.perms[idx][1] = (signed char)b;
        T.perms[idx][2] = (signed char)c; T.perms[idx][3] = (signed char)d;
        ++idx;
      }
    }
  const int Y[6][2] = {{0,1},{0,2},{0,3},{1,2},{1,3},{2,3}};
  int stab[4] = {0,0,0,0};
  int ns = 0;
  for (int p = 0; p < 24; ++p) {            // stabilizer of (0,1), perm order
    int a = T.perms[p][0], b = T.perms[p][1];
    int lo = a < b ? a : b, hi = a < b ? b : a;
    if (lo == 0 && hi == 1 && ns < 4) stab[ns++] = p;
  }
  for (int yi = 0; yi < 6; ++yi) {
    int gy = 0;
    for (int p = 0; p < 24; ++p) {          // first perm with act(p,(0,1))==y
      int a = T.perms[p][0], b = T.perms[p][1];
      int lo = a < b ? a : b, hi = a < b ? b : a;
      if (lo == Y[yi][0] && hi == Y[yi][1]) { gy = p; break; }
    }
    for (int si = 0; si < 4; ++si) {
      int u[4];
      for (int i = 0; i < 4; ++i) u[i] = T.perms[gy][T.perms[stab[si]][i]]; // gy∘s
      for (int i = 0; i < 24; ++i) {
        int ip[4];
        for (int t = 0; t < 4; ++t) ip[T.perms[i][t]] = t;                 // i^-1
        int v[4];
        for (int t = 0; t < 4; ++t) v[t] = u[ip[t]];                       // u∘i^-1
        int vid = 0;
        for (int p = 0; p < 24; ++p) {
          bool eq = true;
          for (int t = 0; t < 4; ++t) if (T.perms[p][t] != v[t]) eq = false;
          if (eq) { vid = p; break; }
        }
        T.q[yi][si][vid] = (signed char)i;                                 // q = p^-1
      }
    }
  }
  return T;
}

__constant__ QTab kT = make_qtab();

// ---------------- WMMA fragment types --------------------------------------
typedef _Float16 v16h __attribute__((ext_vector_type(16)));
typedef _Float16 v8h  __attribute__((ext_vector_type(8)));
typedef float    v8f  __attribute__((ext_vector_type(8)));

// lift_idx[j][d] = sum_u bit_u(d) << (3 - perms[j][u]); bit_u = (d>>(3-u))&1
__device__ __forceinline__ int lift_index(int j, int d) {
  int r = 0;
#pragma unroll
  for (int u = 0; u < 4; ++u)
    r |= ((d >> (3 - u)) & 1) << (3 - (int)kT.perms[j][u]);
  return r;
}

// A-matrix (16-bit, 16x32) slot->K mapping, ISA 7.12.2: lanes0-15 K=0-7,16-23
__device__ __forceinline__ int a_k(int t, int hi) { return (t < 8 ? t : t + 8) + (hi ? 8 : 0); }
// B-matrix (16-bit, 32x16): same slot->K mapping as A: lanes0-15 K=0-7,16-23
__device__ __forceinline__ int b_k(int t, int hi) { return (t < 8 ? t : t + 8) + (hi ? 8 : 0); }

// leaky_relu(x) = max(x, 0.01x): v_mul + v_max, no cmp/cndmask
__device__ __forceinline__ float leaky(float x) { return fmaxf(x, 0.01f * x); }

// ---------------- weight pre-pack into WMMA B-fragment layout ---------------
// W1p: [w][n0][lane][slot]  (24*8*512 halves),  K=16 zero-padded to 32
// W2p: [w][n1][c][lane][slot] (24*4*4*512 halves)
__global__ __launch_bounds__(256)
void pack_weights(const float* __restrict__ W1, const float* __restrict__ W2,
                  _Float16* __restrict__ W1p, _Float16* __restrict__ W2p) {
  int i = blockIdx.x * blockDim.x + threadIdx.x;
  if (i < 24 * 8 * 512) {
    int t = i & 15, lane = (i >> 4) & 31, n0 = (i >> 9) & 7, w = i >> 12;
    int hi = lane >> 4, lm = lane & 15;
    int k = b_k(t, hi);
    float v = W1[(w * 16 + (k & 15)) * 128 + n0 * 16 + lm];
    W1p[i] = (_Float16)(k < 16 ? v : 0.0f);
  }
  if (i < 24 * 16 * 512) {
    int t = i & 15, lane = (i >> 4) & 31, c = (i >> 9) & 3, n1 = (i >> 11) & 3, w = i >> 13;
    int hi = lane >> 4, lm = lane & 15;
    int k = 32 * c + b_k(t, hi);
    W2p[i] = (_Float16)W2[(w * 128 + k) * 64 + n1 * 16 + lm];
  }
}

// ---------------- fused main kernel ----------------------------------------
template <bool PACKED>
__global__ __launch_bounds__(256)
void pe_fc_kernel(const float* __restrict__ x, const float* __restrict__ W1,
                  const float* __restrict__ W2, const float* __restrict__ W3,
                  const _Float16* __restrict__ W1p, const _Float16* __restrict__ W2p,
                  float* __restrict__ out) {
  __shared__ float xs[16];
  __shared__ __align__(16) _Float16 Y1s[8][2048];  // per-wave, col-major [k][m] (128x16)
  __shared__ float Mt[24 * 24];                    // M[j][w] for this b
  __shared__ float part[24];

  const int b    = blockIdx.x;
  const int tid  = threadIdx.x;
  const int wave = tid >> 5;
  const int lane = tid & 31;
  const int lm   = lane & 15;
  const int hi   = lane >> 4;

  if (tid < 16) xs[tid] = x[b * 16 + tid];
  __syncthreads();

  _Float16* Y1 = Y1s[wave];

  for (int tile = 0; tile < 2; ++tile) {
    // ---- A fragment for GEMM1: rows m=lm -> lift j = tile*16+m, K=16 padded to 32
    const int j   = tile * 16 + lm;
    const bool jv = (j < 24);
    const int jc  = jv ? j : 0;
    v16h a1;
#pragma unroll
    for (int t = 0; t < 16; ++t) {
      int k = a_k(t, hi);
      float v = xs[lift_index(jc, k & 15)];       // unconditional LDS read
      a1[t] = (_Float16)((jv && k < 16) ? v : 0.0f);
    }

    for (int wi = 0; wi < 3; ++wi) {
      const int w = wave + 8 * wi;
      const float* W3w = W3 + w * 64;

      // ---- GEMM1: Y1 = leaky(A @ W1[w])  (8 N-tiles of 16) ----
      v16h b1[8];
      if constexpr (PACKED) {
#pragma unroll
        for (int n0 = 0; n0 < 8; ++n0)
          b1[n0] = *(const v16h*)(W1p + (size_t)(w * 8 + n0) * 512 + lane * 16);
      } else {
        const float* W1w = W1 + w * 16 * 128;
#pragma unroll
        for (int n0 = 0; n0 < 8; ++n0)
#pragma unroll
          for (int t = 0; t < 16; ++t) {
            int k = b_k(t, hi);
            float v = W1w[(k & 15) * 128 + n0 * 16 + lm];  // unconditional load
            b1[n0][t] = (_Float16)(k < 16 ? v : 0.0f);     // mask as select
          }
      }
#pragma unroll
      for (int n0 = 0; n0 < 8; ++n0) {
        v8f acc = {};
        acc = __builtin_amdgcn_wmma_f32_16x16x32_f16(false, a1, false, b1[n0],
                                                     (short)0, acc, false, false);
        v8h hv;
#pragma unroll
        for (int v2 = 0; v2 < 8; ++v2) hv[v2] = (_Float16)leaky(acc[v2]);
        // col-major store: col = n0*16+lm, rows hi*8 + [0..7]  -> one b128
        *(v8h*)&Y1[(n0 * 16 + lm) * 16 + hi * 8] = hv;
      }
      asm volatile("s_wait_dscnt 0x0" ::: "memory");  // order DS store -> DS load

      // ---- A fragments for GEMM2: Y1 is 16x128, 4 K-chunks of 32 ----
      v16h a2[4];
#pragma unroll
      for (int c = 0; c < 4; ++c)
#pragma unroll
        for (int t = 0; t < 16; ++t) {
          int k = 32 * c + a_k(t, hi);
          a2[c][t] = Y1[k * 16 + lm];
        }

      // ---- GEMM2 (16x64) fused with GEMM3 (64->1) ----
      float pv[8];
#pragma unroll
      for (int v2 = 0; v2 < 8; ++v2) pv[v2] = 0.0f;
#pragma unroll
      for (int n1 = 0; n1 < 4; ++n1) {
        v16h b2[4];
        if constexpr (PACKED) {
#pragma unroll
          for (int c = 0; c < 4; ++c)
            b2[c] = *(const v16h*)(W2p + (size_t)((w * 4 + n1) * 4 + c) * 512 + lane * 16);
        } else {
          const float* W2w = W2 + w * 128 * 64;
#pragma unroll
          for (int c = 0; c < 4; ++c)
#pragma unroll
            for (int t = 0; t < 16; ++t) {
              int k = 32 * c + b_k(t, hi);
              b2[c][t] = (_Float16)W2w[k * 64 + n1 * 16 + lm];
            }
        }
        v8f acc = {};
#pragma unroll
        for (int c = 0; c < 4; ++c)
          acc = __builtin_amdgcn_wmma_f32_16x16x32_f16(false, a2[c], false, b2[c],
                                                       (short)0, acc, false, false);
        float w3v = W3w[n1 * 16 + lm];        // column n = n1*16+lm of Y2
#pragma unroll
        for (int v2 = 0; v2 < 8; ++v2) pv[v2] += leaky(acc[v2]) * w3v;
      }
      // reduce columns across each 16-lane group (rows m=v / m=v+8)
#pragma unroll
      for (int v2 = 0; v2 < 8; ++v2) {
        float s = pv[v2];
        s += __shfl_xor(s, 1);
        s += __shfl_xor(s, 2);
        s += __shfl_xor(s, 4);
        s += __shfl_xor(s, 8);
        pv[v2] = s;
      }
      if (lm == 0) {
#pragma unroll
        for (int v2 = 0; v2 < 8; ++v2) {
          int jj = tile * 16 + hi * 8 + v2;
          if (jj < 24) Mt[jj * 24 + w] = leaky(pv[v2]);   // M[b, j, w]
        }
      }
    }
  }
  __syncthreads();

  // ---- gather: out[b,y] = mean_s sum_j M[b, j, q[y,s,j]] ----
  if (tid < 24) {
    int y = tid >> 2, s = tid & 3;
    float acc = 0.0f;
#pragma unroll
    for (int jj = 0; jj < 24; ++jj)
      acc += Mt[jj * 24 + (int)kT.q[y][s][jj]];
    part[tid] = acc;
  }
  __syncthreads();
  if (tid < 6)
    out[b * 6 + tid] =
        0.25f * (part[tid * 4] + part[tid * 4 + 1] + part[tid * 4 + 2] + part[tid * 4 + 3]);
}

extern "C" void kernel_launch(void* const* d_in, const int* in_sizes, int n_in,
                              void* d_out, int out_size, void* d_ws, size_t ws_size,
                              hipStream_t stream) {
  const float* x  = (const float*)d_in[0];   // (B,16)
  const float* W1 = (const float*)d_in[1];   // (24,16,128)
  const float* W2 = (const float*)d_in[2];   // (24,128,64)
  const float* W3 = (const float*)d_in[3];   // (24,64,1)
  float* out = (float*)d_out;                // (B,6)
  const int B = in_sizes[0] / 16;

  const size_t n_w1p = (size_t)24 * 8 * 512;    //  98304 halves
  const size_t n_w2p = (size_t)24 * 16 * 512;   // 196608 halves
  const size_t need  = (n_w1p + n_w2p) * 2;     // bytes

  if (d_ws != nullptr && ws_size >= need) {
    _Float16* W1p = (_Float16*)d_ws;
    _Float16* W2p = W1p + n_w1p;
    const int n = (int)(24 * 16 * 512);
    pack_weights<<<(n + 255) / 256, 256, 0, stream>>>(W1, W2, W1p, W2p);
    pe_fc_kernel<true><<<B, 256, 0, stream>>>(x, W1, W2, W3, W1p, W2p, out);
  } else {
    pe_fc_kernel<false><<<B, 256, 0, stream>>>(x, W1, W2, W3, nullptr, nullptr, out);
  }
}
